// SlowNeuron_55396488183955
// MI455X (gfx1250) — compile-verified, matrix operand order checked by reading, not run on hardware
//
#include <hip/hip_runtime.h>
#include <math.h>

// ---------------------------------------------------------------------------
// SlowNeuron for MI455X (gfx1250, wave32).
// Memory-bound (~900 MB HBM traffic @ 23.3 TB/s => ~40us floor).
// GEMMs: v_wmma_f32_16x16x32_f16, one wave per 16x16 D-tile, K padded to 32.
// A and B^T staged in f16 so each K-step is four 16-byte loads + 1 WMMA,
// branch-free. Fast-weight rank-1 updates are float4 streaming kernels.
// ---------------------------------------------------------------------------

typedef __attribute__((ext_vector_type(16))) _Float16 v16h;
typedef __attribute__((ext_vector_type(8)))  _Float16 v8h;
typedef __attribute__((ext_vector_type(8)))  float    v8f;

#define BB   64
#define TT   8
#define DMD  256
#define RR   32
#define DQD  64
#define DVD  128
#define DPD  64
#define DCD  128
#define DHD  192
#define FDMD 2560
#define FDPD 128
#define NFD  2

__device__ __forceinline__ float geluf(float x)     { return 0.5f * x * (1.0f + erff(x * 0.70710678118654752f)); }
__device__ __forceinline__ float sigmoidf_(float x) { return 1.0f / (1.0f + expf(-x)); }
__device__ __forceinline__ float softplusf_(float x){ return x > 20.0f ? x : log1pf(expf(x)); }

// ---------------------------------------------------------------------------
// WMMA GEMM: D[M,N] = act(A[M,Kp] @ B^T[N,Kp] + bias[N]), f16 operands, f32 acc.
// Kp is a multiple of 32, zero padded => no bounds checks in the K loop.
// Fragment mapping (16-bit A 16x32): lane L holds row M=L%16; element e<8 is
// k=khalf+e, e>=8 is k=16+khalf+(e-8), khalf=(L/16)*8 -> two 8-half runs.
// B^T rows give the identical two contiguous runs for the B fragment.
// C/D: VGPR j holds M=j+(L/16)*8, N=L%16.
// act: 0=none, 1=exact GELU. D (f32) and Dh (f16, stride N) each optional.
// ---------------------------------------------------------------------------
__global__ void k_gemm_wmma(const _Float16* __restrict__ Ah, const _Float16* __restrict__ Bh,
                            const float* __restrict__ bias, float* __restrict__ D,
                            _Float16* __restrict__ Dh, int M, int Kp, int N, int act)
{
    const int lane   = threadIdx.x;            // 0..31, one wave per block
    const int ntiles = N >> 4;
    const int tile   = blockIdx.x;
    const int m0     = (tile / ntiles) << 4;
    const int n0     = (tile % ntiles) << 4;
    const int khalf  = ((lane >> 4) & 1) * 8;

    const _Float16* ap = Ah + (size_t)(m0 + (lane & 15)) * Kp + khalf;
    const _Float16* bp = Bh + (size_t)(n0 + (lane & 15)) * Kp + khalf;

    v8f acc = {};
    for (int k0 = 0; k0 < Kp; k0 += 32) {
        v8h a0 = *(const v8h*)(ap + k0);
        v8h a1 = *(const v8h*)(ap + k0 + 16);
        v8h b0 = *(const v8h*)(bp + k0);
        v8h b1 = *(const v8h*)(bp + k0 + 16);
        v16h a = __builtin_shufflevector(a0, a1, 0,1,2,3,4,5,6,7,8,9,10,11,12,13,14,15);
        v16h b = __builtin_shufflevector(b0, b1, 0,1,2,3,4,5,6,7,8,9,10,11,12,13,14,15);
        acc = __builtin_amdgcn_wmma_f32_16x16x32_f16(
                  false, a, false, b, (short)0, acc, false, false);
    }
    const int mbase = m0 + ((lane >> 4) & 1) * 8;
    const int n     = n0 + (lane & 15);
    const float bn  = bias ? bias[n] : 0.0f;
#pragma unroll
    for (int j = 0; j < 8; ++j) {
        float r = acc[j] + bn;
        if (act == 1) r = geluf(r);
        if (D)  D[(size_t)(mbase + j) * N + n]  = r;
        if (Dh) Dh[(size_t)(mbase + j) * N + n] = (_Float16)r;
    }
}

// weight pre-pack: W[K,N] f32 -> BT[N,Kp] f16, zero padded
__global__ void k_packB(const float* __restrict__ W, _Float16* __restrict__ BT,
                        int K, int N, int Kp)
{
    int idx = blockIdx.x * blockDim.x + threadIdx.x;
    if (idx >= N * Kp) return;
    int n = idx / Kp, k = idx % Kp;
    BT[idx] = (k < K) ? (_Float16)W[(size_t)k * N + n] : (_Float16)0.0f;
}

// out[row] = post(dot(X[row,:K], w) + bias[0]); mode 0: sigmoid, 1: min(softplus, cap)
__global__ void k_rowdot(const float* __restrict__ X, int K, const float* __restrict__ w,
                         const float* __restrict__ bias, float* __restrict__ out,
                         int mode, float cap)
{
    const int row = blockIdx.x, lane = threadIdx.x;
    float s = 0.0f;
    for (int k = lane; k < K; k += 32) s += X[(size_t)row * K + k] * w[k];
    for (int off = 16; off > 0; off >>= 1) s += __shfl_down(s, off, 32);
    if (lane == 0) {
        float r = s + bias[0];
        r = (mode == 0) ? sigmoidf_(r) : fminf(softplusf_(r), cap);
        out[row] = r;
    }
}

// mean over T -> h_avg (f32); xsur16[b,:256]=h_avg-prev, [256:384]=ctx (f16)
__global__ void k_avg_pack(const float* __restrict__ reports, const float* __restrict__ prev,
                           const float* __restrict__ dctx, float* __restrict__ h_avg,
                           _Float16* __restrict__ xsur16)
{
    int idx = blockIdx.x * blockDim.x + threadIdx.x;
    if (idx >= BB * 384) return;
    int b = idx / 384, c = idx % 384;
    if (c < DMD) {
        float s = 0.0f;
        for (int t = 0; t < TT; ++t) s += reports[((size_t)t * BB + b) * DMD + c];
        s *= (1.0f / TT);
        h_avg[b * DMD + c] = s;
        xsur16[idx] = (_Float16)(s - prev[b * DMD + c]);
    } else {
        xsur16[idx] = (_Float16)dctx[c - DMD];
    }
}

// wi16[b, 0:416]: h_avg | surprise | ctx | zero-pad   (Kp=416)
__global__ void k_pack_wi(const float* __restrict__ h_avg, const float* __restrict__ sur,
                          const float* __restrict__ dctx, _Float16* __restrict__ wi16)
{
    int idx = blockIdx.x * blockDim.x + threadIdx.x;
    if (idx >= BB * 416) return;
    int b = idx / 416, c = idx % 416;
    float v;
    if (c < DMD)       v = h_avg[b * DMD + c];
    else if (c == DMD) v = sur[b];
    else if (c < 385)  v = dctx[c - DMD - 1];
    else               v = 0.0f;
    wi16[idx] = (_Float16)v;
}

// xlr16[b, 0:160]: surprise | ctx | zero-pad   (Kp=160)
__global__ void k_pack_xlr(const float* __restrict__ sur, const float* __restrict__ dctx,
                           _Float16* __restrict__ xlr16)
{
    int idx = blockIdx.x * blockDim.x + threadIdx.x;
    if (idx >= BB * 160) return;
    int b = idx / 160, c = idx % 160;
    float v;
    if (c == 0)       v = sur[b];
    else if (c < 129) v = dctx[c - 1];
    else              v = 0.0f;
    xlr16[idx] = (_Float16)v;
}

// xq16[b, 0:384]: h_avg | ctx
__global__ void k_pack_xq(const float* __restrict__ h_avg, const float* __restrict__ dctx,
                          _Float16* __restrict__ xq16)
{
    int idx = blockIdx.x * blockDim.x + threadIdx.x;
    if (idx >= BB * 384) return;
    int b = idx / 384, c = idx % 384;
    xq16[idx] = (_Float16)((c < DMD) ? h_avg[b * DMD + c] : dctx[c - DMD]);
}

// xpw16[b, 0:672]: h_avg | surprise | mem_read | ctx | zero-pad  (Kp=672)
__global__ void k_pack_xpw(const float* __restrict__ h_avg, const float* __restrict__ sur,
                           const float* __restrict__ mem_read, const float* __restrict__ dctx,
                           _Float16* __restrict__ xpw16)
{
    int idx = blockIdx.x * blockDim.x + threadIdx.x;
    if (idx >= BB * 672) return;
    int b = idx / 672, c = idx % 672;
    float v;
    if (c < 256)       v = h_avg[b * DMD + c];
    else if (c == 256) v = sur[b];
    else if (c < 513)  v = mem_read[b * DMD + (c - 257)];
    else if (c < 641)  v = dctx[c - 513];
    else               v = 0.0f;
    xpw16[idx] = (_Float16)v;
}

// memA = mem_A + lr * v (x) k ; per-batch Frobenius sumsq -> bnorm
__global__ void k_memA(const float* __restrict__ mem_A, const float* __restrict__ lr,
                       const float* __restrict__ v, const float* __restrict__ kk,
                       float* __restrict__ memA_out, float* __restrict__ bnorm)
{
    const int b = blockIdx.x, t = threadIdx.x;
    const float lrb = lr[b];
    float ss = 0.0f;
    for (int i = t; i < DMD * RR; i += 256) {
        int m = i >> 5, r = i & 31;
        float val = mem_A[(size_t)b * DMD * RR + i] + lrb * v[b * DMD + m] * kk[b * RR + r];
        memA_out[(size_t)b * DMD * RR + i] = val;
        ss += val * val;
    }
    __shared__ float red[256];
    red[t] = ss; __syncthreads();
    for (int s = 128; s > 0; s >>= 1) { if (t < s) red[t] += red[t + s]; __syncthreads(); }
    if (t == 0) bnorm[b] = sqrtf(red[0]);
}

// clip-scale memA in place and transpose into slots16[(b*R+r)*DM + m] (f16)
__global__ void k_memA_fin(float* __restrict__ memA_out, const float* __restrict__ bnorm,
                           _Float16* __restrict__ slots16)
{
    int idx = blockIdx.x * blockDim.x + threadIdx.x;
    if (idx >= BB * DMD * RR) return;
    int b = idx >> 13, i = idx & 8191;
    int m = i >> 5, r = i & 31;
    float nrm = bnorm[b];
    float sc  = (nrm > 10.0f) ? (10.0f / (nrm + 1e-8f)) : 1.0f;
    float val = memA_out[idx] * sc;
    memA_out[idx] = val;
    slots16[((size_t)b * RR + r) * DMD + m] = (_Float16)val;
}

// scores -> softmax over R=32 -> mrc16[b, DV] (f16)
__global__ void k_attn(const float* __restrict__ q, const float* __restrict__ keys,
                       const float* __restrict__ vals, _Float16* __restrict__ mrc16)
{
    const int b = blockIdx.x, t = threadIdx.x;   // block = 128
    __shared__ float aw[RR];
    if (t < RR) {
        const float* kr = keys + ((size_t)b * RR + t) * DQD;
        const float* qb = q + b * DQD;
        float s = 0.0f;
        for (int i = 0; i < DQD; ++i) s += qb[i] * kr[i];
        aw[t] = s * 0.125f;                      // 1/sqrt(DQ)
    }
    __syncthreads();
    if (t == 0) {
        float mx = -1e30f;
        for (int r = 0; r < RR; ++r) mx = fmaxf(mx, aw[r]);
        float sm = 0.0f;
        for (int r = 0; r < RR; ++r) { aw[r] = expf(aw[r] - mx); sm += aw[r]; }
        float inv = 1.0f / sm;
        for (int r = 0; r < RR; ++r) aw[r] *= inv;
    }
    __syncthreads();
    float s = 0.0f;
    for (int r = 0; r < RR; ++r) s += aw[r] * vals[((size_t)b * RR + r) * DVD + t];
    mrc16[b * DVD + t] = (_Float16)s;
}

// per-batch low-rank adapter: slow = mem_read + gelu(mr @ Wd) @ Wu  (f32 + f16 copy)
__global__ void k_slow(const float* __restrict__ mem_read, const float* __restrict__ Wdb,
                       const float* __restrict__ Wdm, const float* __restrict__ Wub,
                       const float* __restrict__ Wum, float* __restrict__ slow,
                       _Float16* __restrict__ slow16)
{
    const int b = blockIdx.x, t = threadIdx.x;   // block = 256
    __shared__ float mr[DMD];
    __shared__ float down[DPD];
    mr[t] = mem_read[b * DMD + t];
    __syncthreads();
    if (t < DPD) {
        float s = 0.0f;
        const float* wm = Wdm + (size_t)b * DMD * DPD;
        for (int m = 0; m < DMD; ++m) s += mr[m] * (Wdb[m * DPD + t] + wm[m * DPD + t]);
        down[t] = geluf(s);
    }
    __syncthreads();
    float s = 0.0f;
    const float* wu = Wum + (size_t)b * DPD * DMD;
    for (int p = 0; p < DPD; ++p) s += down[p] * (Wub[p * DMD + t] + wu[p * DMD + t]);
    float r = mr[t] + s;
    slow[b * DMD + t] = r;
    slow16[b * DMD + t] = (_Float16)r;
}

// thr = sigmoid(ctx . th_w + th_b); gg[b] = surprise[b] > thr ? plr[b] : 0
__global__ void k_gg(const float* __restrict__ dctx, const float* __restrict__ th_w,
                     const float* __restrict__ th_b, const float* __restrict__ sur,
                     const float* __restrict__ plr, float* __restrict__ gg)
{
    const int t = threadIdx.x;                   // block = 128
    __shared__ float red[DCD];
    __shared__ float thr;
    red[t] = dctx[t] * th_w[t];
    __syncthreads();
    for (int s = 64; s > 0; s >>= 1) { if (t < s) red[t] += red[t + s]; __syncthreads(); }
    if (t == 0) thr = sigmoidf_(red[0] + th_b[0]);
    __syncthreads();
    if (t < BB) gg[t] = (sur[t] > thr) ? plr[t] : 0.0f;
}

// Wdm/Wum rank-1 updates
__global__ void k_wmod(const float* __restrict__ Wdm_in, const float* __restrict__ Wum_in,
                       const float* __restrict__ raw, const float* __restrict__ gg,
                       float* __restrict__ Wdm_out, float* __restrict__ Wum_out)
{
    int idx = blockIdx.x * blockDim.x + threadIdx.x;
    const int half = BB * DMD * DPD;             // 1,048,576
    if (idx < half) {
        int b = idx >> 14, m = (idx >> 6) & 255, p = idx & 63;
        Wdm_out[idx] = Wdm_in[idx] + gg[b] * raw[b * 640 + m] * raw[b * 640 + 256 + p];
    } else if (idx < 2 * half) {
        int j = idx - half;
        int b = j >> 14, p = (j >> 8) & 63, m = j & 255;
        Wum_out[j] = Wum_in[j] + gg[b] * raw[b * 640 + 320 + p] * raw[b * 640 + 384 + m];
    }
}

// Frobenius norms of fast_W_{down,up}_mod per (n,b): 327,680 elems each, float4.
__global__ void k_fastnorm(const float* __restrict__ fdn_in, const float* __restrict__ fun_in,
                           float* __restrict__ dn, float* __restrict__ un)
{
    const int id = blockIdx.x;                   // 0..255
    const int which = id >> 7, nb = id & 127;
    const float4* b4 = (const float4*)((which ? fun_in : fdn_in) + (size_t)nb * FDMD * FDPD);
    float s = 0.0f;
    for (int i = threadIdx.x; i < (FDMD * FDPD) / 4; i += 256) {
        float4 x = b4[i];
        s += x.x * x.x + x.y * x.y + x.z * x.z + x.w * x.w;
    }
    __shared__ float red[256];
    red[threadIdx.x] = s; __syncthreads();
    for (int st = 128; st > 0; st >>= 1) { if (threadIdx.x < st) red[threadIdx.x] += red[threadIdx.x + st]; __syncthreads(); }
    if (threadIdx.x == 0) (which ? un : dn)[nb] = sqrtf(red[0]);
}

// ci16[n,b, 0:288]: slow | dn | un | zero-pad  (Kp=288)
__global__ void k_pack_ci(const float* __restrict__ slow, const float* __restrict__ dn,
                          const float* __restrict__ un, _Float16* __restrict__ ci16)
{
    int idx = blockIdx.x * blockDim.x + threadIdx.x;
    if (idx >= NFD * BB * 288) return;
    int n = idx / (BB * 288), r = idx % (BB * 288);
    int b = r / 288, c = r % 288;
    float v;
    if (c < DMD)       v = slow[b * DMD + c];
    else if (c == DMD) v = dn[n * BB + b];
    else if (c == 257) v = un[n * BB + b];
    else               v = 0.0f;
    ci16[idx] = (_Float16)v;
}

// normalize the 4 slices of each craw row in place (L2, clamped at 1e-12)
__global__ void k_crawnorm(float* __restrict__ craw)
{
    const int nb = blockIdx.x;                   // 0..127
    float* row = craw + (size_t)nb * 5376;
    __shared__ float red[256];
    __shared__ float norms[4];
    const int off[4] = {0, 2560, 2688, 2816};
    const int len[4] = {2560, 128, 128, 2560};
    for (int s = 0; s < 4; ++s) {
        float ss = 0.0f;
        for (int i = threadIdx.x; i < len[s]; i += 256) { float v = row[off[s] + i]; ss += v * v; }
        red[threadIdx.x] = ss; __syncthreads();
        for (int st = 128; st > 0; st >>= 1) { if (threadIdx.x < st) red[threadIdx.x] += red[threadIdx.x + st]; __syncthreads(); }
        if (threadIdx.x == 0) norms[s] = fmaxf(sqrtf(red[0]), 1e-12f);
        __syncthreads();
    }
    for (int s = 0; s < 4; ++s) {
        float inv = 1.0f / norms[s];
        for (int i = threadIdx.x; i < len[s]; i += 256) row[off[s] + i] *= inv;
    }
}

// fdn[n,b,i,j] = fast_dn + clr[b] * cd_pat[i] * cd_addr[j]   (float4 over j, 32/row)
__global__ void k_fast_dn(const float4* __restrict__ in, const float* __restrict__ craw,
                          const float* __restrict__ clr, float4* __restrict__ out)
{
    size_t e = (size_t)blockIdx.x * blockDim.x + threadIdx.x;
    int j4 = (int)(e & 31);
    size_t row = e >> 5;                         // nb*FDM + i
    int ii = (int)(row % FDMD);
    size_t nb = row / FDMD;
    int b = (int)(nb & 63);
    const float* cr = craw + nb * 5376;
    float g = clr[b] * cr[ii];
    float4 ad = ((const float4*)(cr + FDMD))[j4];
    float4 x = in[e];
    float4 r = {x.x + g * ad.x, x.y + g * ad.y, x.z + g * ad.z, x.w + g * ad.w};
    out[e] = r;
}

// fun[n,b,p,m] = fast_up + clr[b] * cu_pat[p] * cu_addr[m]   (float4 over m, 640/row)
__global__ void k_fast_up(const float4* __restrict__ in, const float* __restrict__ craw,
                          const float* __restrict__ clr, float4* __restrict__ out)
{
    size_t e = (size_t)blockIdx.x * blockDim.x + threadIdx.x;
    int j4 = (int)(e % 640);
    size_t row = e / 640;                        // nb*FDP + p
    int p = (int)(row % FDPD);
    size_t nb = row / FDPD;
    int b = (int)(nb & 63);
    const float* cr = craw + nb * 5376;
    float g = clr[b] * cr[2688 + p];
    float4 ad = ((const float4*)(cr + 2816))[j4];
    float4 x = in[e];
    float4 r = {x.x + g * ad.x, x.y + g * ad.y, x.z + g * ad.z, x.w + g * ad.w};
    out[e] = r;
}

// ---------------------------------------------------------------------------
static inline int cdiv(int a, int b) { return (a + b - 1) / b; }

static inline void gemm(const _Float16* Ah, const _Float16* Bh, const float* bias,
                        float* D, _Float16* Dh, int M, int Kp, int N, int act,
                        hipStream_t s)
{
    k_gemm_wmma<<<dim3((M / 16) * (N / 16)), dim3(32), 0, s>>>(Ah, Bh, bias, D, Dh, M, Kp, N, act);
}

extern "C" void kernel_launch(void* const* d_in, const int* in_sizes, int n_in,
                              void* d_out, int out_size, void* d_ws, size_t ws_size,
                              hipStream_t stream)
{
    const float* reports   = (const float*)d_in[0];
    const float* mem_A     = (const float*)d_in[1];
    const float* W_dmod    = (const float*)d_in[2];
    const float* W_umod    = (const float*)d_in[3];
    const float* prev_mr   = (const float*)d_in[4];
    const float* fast_dn_i = (const float*)d_in[5];
    const float* fast_un_i = (const float*)d_in[6];
    const float* dctx      = (const float*)d_in[7];
    const float* sur_w1 = (const float*)d_in[8],  *sur_b1 = (const float*)d_in[9];
    const float* sur_w2 = (const float*)d_in[10], *sur_b2 = (const float*)d_in[11];
    const float* wk_w1  = (const float*)d_in[12], *wk_b1  = (const float*)d_in[13];
    const float* wk_w2  = (const float*)d_in[14], *wk_b2  = (const float*)d_in[15];
    const float* wv_w1  = (const float*)d_in[16], *wv_b1  = (const float*)d_in[17];
    const float* wv_w2  = (const float*)d_in[18], *wv_b2  = (const float*)d_in[19];
    const float* lr_w1  = (const float*)d_in[20], *lr_b1  = (const float*)d_in[21];
    const float* lr_w2  = (const float*)d_in[22], *lr_b2  = (const float*)d_in[23];
    const float* rq_w   = (const float*)d_in[24], *rq_b   = (const float*)d_in[25];
    const float* W_K    = (const float*)d_in[26], *W_V    = (const float*)d_in[27];
    const float* vup_w  = (const float*)d_in[28], *vup_b  = (const float*)d_in[29];
    const float* W_dbase = (const float*)d_in[30], *W_ubase = (const float*)d_in[31];
    const float* th_w   = (const float*)d_in[32], *th_b   = (const float*)d_in[33];
    const float* pw_w1  = (const float*)d_in[34], *pw_b1  = (const float*)d_in[35];
    const float* pw_w2  = (const float*)d_in[36], *pw_b2  = (const float*)d_in[37];
    const float* plr_w1 = (const float*)d_in[38], *plr_b1 = (const float*)d_in[39];
    const float* plr_w2 = (const float*)d_in[40], *plr_b2 = (const float*)d_in[41];
    const float* cc_w   = (const float*)d_in[42], *cc_b   = (const float*)d_in[43];
    const float* cons_w1 = (const float*)d_in[44], *cons_b1 = (const float*)d_in[45];
    const float* cons_w2 = (const float*)d_in[46], *cons_b2 = (const float*)d_in[47];
    const float* clr_w  = (const float*)d_in[48], *clr_b  = (const float*)d_in[49];

    // output segments (flat, return order)
    float* out      = (float*)d_out;
    float* o_ctx    = out;                                   // 64*128
    float* o_memA   = out + 8192;                            // 64*256*32
    float* o_Wdm    = o_memA + (size_t)BB * DMD * RR;        // 64*256*64
    float* o_Wum    = o_Wdm + (size_t)BB * DMD * DPD;
    float* o_fdn    = o_Wum + (size_t)BB * DPD * DMD;        // 2*64*2560*128
    float* o_fun    = o_fdn + (size_t)NFD * BB * FDMD * FDPD;

    // ---- workspace: f32 region then f16 region (16B aligned) ----
    float* wf = (float*)d_ws;
    size_t o = 0;
    auto alloc = [&](size_t n) { float* p = wf + o; o += n; return p; };
    float* h_avg   = alloc(BB * DMD);
    float* hid_sur = alloc(BB * DHD);
    float* surprise= alloc(BB);
    float* hid_lr  = alloc(BB * 96);
    float* lrb     = alloc(BB);
    float* bnorm   = alloc(BB);
    float* qbuf    = alloc(BB * DQD);
    float* kbuf    = alloc(BB * RR);
    float* vbuf    = alloc(BB * DMD);
    float* keys    = alloc((size_t)BB * RR * DQD);
    float* vals    = alloc((size_t)BB * RR * DVD);
    float* mem_read= alloc(BB * DMD);
    float* slow    = alloc(BB * DMD);
    float* raw     = alloc(BB * 640);
    float* hid_plr = alloc(BB * 32);
    float* plr     = alloc(BB);
    float* gg      = alloc(BB);
    float* clr     = alloc(BB);
    float* dn      = alloc(NFD * BB);
    float* un      = alloc(NFD * BB);
    float* craw    = alloc((size_t)NFD * BB * 5376);

    _Float16* wh = (_Float16*)(wf + o);
    size_t oh = 0;
    auto alloc16 = [&](size_t n) { _Float16* p = wh + oh; oh += n; return p; };
    // activations (f16, Kp-padded rows)
    _Float16* xsur16  = alloc16(BB * 384);
    _Float16* wi16    = alloc16(BB * 416);
    _Float16* xlr16   = alloc16(BB * 160);
    _Float16* xq16    = alloc16(BB * 384);
    _Float16* xpw16   = alloc16(BB * 672);
    _Float16* ci16    = alloc16((size_t)NFD * BB * 288);
    _Float16* hid_k16 = alloc16(BB * DHD);
    _Float16* hid_v16 = alloc16(BB * DHD);
    _Float16* hid_pw16= alloc16(BB * DHD);
    _Float16* ch16    = alloc16((size_t)NFD * BB * DHD);
    _Float16* mrc16   = alloc16(BB * DVD);
    _Float16* slots16 = alloc16((size_t)BB * RR * DMD);
    _Float16* slow16  = alloc16(BB * DMD);
    // weights (f16, transposed [N, Kp])
    _Float16* B_sur1  = alloc16((size_t)192 * 384);
    _Float16* B_wk1   = alloc16((size_t)192 * 416);
    _Float16* B_wk2   = alloc16((size_t)32 * 192);
    _Float16* B_wv1   = alloc16((size_t)192 * 416);
    _Float16* B_wv2   = alloc16((size_t)256 * 192);
    _Float16* B_lr1   = alloc16((size_t)96 * 160);
    _Float16* B_rq    = alloc16((size_t)64 * 384);
    _Float16* B_WK    = alloc16((size_t)64 * 256);
    _Float16* B_WV    = alloc16((size_t)128 * 256);
    _Float16* B_vup   = alloc16((size_t)256 * 128);
    _Float16* B_pw1   = alloc16((size_t)192 * 672);
    _Float16* B_pw2   = alloc16((size_t)640 * 192);
    _Float16* B_plr1  = alloc16((size_t)32 * 160);
    _Float16* B_cc    = alloc16((size_t)128 * 256);
    _Float16* B_cons1 = alloc16((size_t)NFD * 192 * 288);
    _Float16* B_cons2 = alloc16((size_t)NFD * 5376 * 192);
    (void)ws_size; (void)in_sizes; (void)n_in; (void)out_size;

    auto packB = [&](const float* W, _Float16* BT, int K, int N, int Kp) {
        k_packB<<<cdiv(N * Kp, 256), 256, 0, stream>>>(W, BT, K, N, Kp);
    };

    // 0. weight pre-pack (f16, transposed, K zero-padded to multiple of 32)
    packB(sur_w1, B_sur1, 384, 192, 384);
    packB(wk_w1,  B_wk1,  385, 192, 416);
    packB(wk_w2,  B_wk2,  192, 32,  192);
    packB(wv_w1,  B_wv1,  385, 192, 416);
    packB(wv_w2,  B_wv2,  192, 256, 192);
    packB(lr_w1,  B_lr1,  129, 96,  160);
    packB(rq_w,   B_rq,   384, 64,  384);
    packB(W_K,    B_WK,   256, 64,  256);
    packB(W_V,    B_WV,   256, 128, 256);
    packB(vup_w,  B_vup,  128, 256, 128);
    packB(pw_w1,  B_pw1,  641, 192, 672);
    packB(pw_w2,  B_pw2,  192, 640, 192);
    packB(plr_w1, B_plr1, 129, 32,  160);
    packB(cc_w,   B_cc,   256, 128, 256);
    for (int n = 0; n < NFD; ++n) {
        packB(cons_w1 + (size_t)n * 258 * 192,  B_cons1 + (size_t)n * 192 * 288,  258, 192,  288);
        packB(cons_w2 + (size_t)n * 192 * 5376, B_cons2 + (size_t)n * 5376 * 192, 192, 5376, 192);
    }

    // 1. h_avg, err|ctx pack
    k_avg_pack<<<cdiv(BB * 384, 256), 256, 0, stream>>>(reports, prev_mr, dctx, h_avg, xsur16);
    // 2. surprise MLP
    gemm(xsur16, B_sur1, sur_b1, hid_sur, nullptr, BB, 384, DHD, 1, stream);
    k_rowdot<<<BB, 32, 0, stream>>>(hid_sur, DHD, sur_w2, sur_b2, surprise, 0, 0.0f);
    // 3. packs
    k_pack_wi<<<cdiv(BB * 416, 256), 256, 0, stream>>>(h_avg, surprise, dctx, wi16);
    k_pack_xlr<<<cdiv(BB * 160, 256), 256, 0, stream>>>(surprise, dctx, xlr16);
    k_pack_xq<<<cdiv(BB * 384, 256), 256, 0, stream>>>(h_avg, dctx, xq16);
    // 4. k / v MLPs
    gemm(wi16, B_wk1, wk_b1, nullptr, hid_k16, BB, 416, DHD, 1, stream);
    gemm(hid_k16, B_wk2, wk_b2, kbuf, nullptr, BB, DHD, RR, 0, stream);
    gemm(wi16, B_wv1, wv_b1, nullptr, hid_v16, BB, 416, DHD, 1, stream);
    gemm(hid_v16, B_wv2, wv_b2, vbuf, nullptr, BB, DHD, DMD, 0, stream);
    // 5. lr
    gemm(xlr16, B_lr1, lr_b1, hid_lr, nullptr, BB, 160, 96, 1, stream);
    k_rowdot<<<BB, 32, 0, stream>>>(hid_lr, 96, lr_w2, lr_b2, lrb, 1, 0.1f);
    // 6. memA update + clip + transpose to slots (f16)
    k_memA<<<BB, 256, 0, stream>>>(mem_A, lrb, vbuf, kbuf, o_memA, bnorm);
    k_memA_fin<<<cdiv(BB * DMD * RR, 256), 256, 0, stream>>>(o_memA, bnorm, slots16);
    // 7. attention read
    gemm(xq16, B_rq, rq_b, qbuf, nullptr, BB, 384, DQD, 0, stream);
    gemm(slots16, B_WK, nullptr, keys, nullptr, BB * RR, DMD, DQD, 0, stream);
    gemm(slots16, B_WV, nullptr, vals, nullptr, BB * RR, DMD, DVD, 0, stream);
    k_attn<<<BB, 128, 0, stream>>>(qbuf, keys, vals, mrc16);
    gemm(mrc16, B_vup, vup_b, mem_read, nullptr, BB, DVD, DMD, 0, stream);
    // 8. per-batch low-rank adapter
    k_slow<<<BB, 256, 0, stream>>>(mem_read, W_dbase, W_dmod, W_ubase, W_umod, slow, slow16);
    // 9. pw MLP
    k_pack_xpw<<<cdiv(BB * 672, 256), 256, 0, stream>>>(h_avg, surprise, mem_read, dctx, xpw16);
    gemm(xpw16, B_pw1, pw_b1, nullptr, hid_pw16, BB, 672, DHD, 1, stream);
    gemm(hid_pw16, B_pw2, pw_b2, raw, nullptr, BB, DHD, 640, 0, stream);
    // 10. plr, gate
    gemm(xlr16, B_plr1, plr_b1, hid_plr, nullptr, BB, 160, 32, 1, stream);
    k_rowdot<<<BB, 32, 0, stream>>>(hid_plr, 32, plr_w2, plr_b2, plr, 1, 0.1f);
    k_gg<<<1, 128, 0, stream>>>(dctx, th_w, th_b, surprise, plr, gg);
    // 11. Wdm / Wum outputs
    k_wmod<<<cdiv(2 * BB * DMD * DPD, 256), 256, 0, stream>>>(W_dmod, W_umod, raw, gg, o_Wdm, o_Wum);
    // 12. new_ctx (direct to output) and clr
    gemm(slow16, B_cc, cc_b, o_ctx, nullptr, BB, DMD, DCD, 0, stream);
    k_rowdot<<<BB, 32, 0, stream>>>(slow, DMD, clr_w, clr_b, clr, 1, 0.01f);
    // 13. fast-weight norms + consolidation MLP
    k_fastnorm<<<2 * NFD * BB, 256, 0, stream>>>(fast_dn_i, fast_un_i, dn, un);
    k_pack_ci<<<cdiv(NFD * BB * 288, 256), 256, 0, stream>>>(slow, dn, un, ci16);
    for (int n = 0; n < NFD; ++n) {
        gemm(ci16 + (size_t)n * BB * 288, B_cons1 + (size_t)n * 192 * 288,
             cons_b1 + (size_t)n * DHD, nullptr, ch16 + (size_t)n * BB * DHD,
             BB, 288, DHD, 1, stream);
        gemm(ch16 + (size_t)n * BB * DHD, B_cons2 + (size_t)n * 5376 * 192,
             cons_b2 + (size_t)n * 5376, craw + (size_t)n * BB * 5376, nullptr,
             BB, DHD, 5376, 0, stream);
    }
    k_crawnorm<<<NFD * BB, 256, 0, stream>>>(craw);
    // 14. streaming fast-weight rank-1 updates (dominant cost, pure HBM bandwidth)
    const size_t n4 = (size_t)NFD * BB * FDMD * FDPD / 4;    // 10,485,760 float4
    k_fast_dn<<<(unsigned)(n4 / 256), 256, 0, stream>>>((const float4*)fast_dn_i, craw, clr, (float4*)o_fdn);
    k_fast_up<<<(unsigned)(n4 / 256), 256, 0, stream>>>((const float4*)fast_un_i, craw, clr, (float4*)o_fun);
}
